// Deimv2LQE_35974646071468
// MI455X (gfx1250) — compile-verified, matrix operand order checked by reading, not run on hardware
//
#include <hip/hip_runtime.h>
#include <hip/hip_bf16.h>
#include <math.h>

typedef __attribute__((ext_vector_type(16))) _Float16 v16h;
typedef __attribute__((ext_vector_type(8)))  float    v8f;

#define NBINS1 33          // bins per corner
#define INDIM  20          // 4 * (TOPK + 1)
#define HID    64
#define CDIM   132         // 4 * 33 floats per token (= 33 float4s)
#define CDIM4  33          // float4s per token
#define TOK_PER_WAVE 16
#define WAVES  4
#define TPB    (WAVES * 32)
#define TOK_PER_BLOCK (WAVES * TOK_PER_WAVE)   // 64
#define LDS_STRIDE 140     // dwords; 560B rows: 16B-aligned for b128, conflict-free reads

union V16H { _Float16 h[16]; v16h v; };

__global__ __launch_bounds__(TPB)
void deimv2_lqe_kernel(const float* __restrict__ scores,
                       const float* __restrict__ pred,
                       const float* __restrict__ w1,
                       const float* __restrict__ b1,
                       const float* __restrict__ w2,
                       const float* __restrict__ b2,
                       float* __restrict__ out,
                       int nTok)
{
    __shared__ __align__(16) float    predLds[WAVES][TOK_PER_WAVE * LDS_STRIDE];
    __shared__ __align__(16) _Float16 statLds[WAVES][TOK_PER_WAVE][32];

    const int lane = threadIdx.x & 31;
    const int w    = threadIdx.x >> 5;
    const int col  = lane & 15;   // M (A/token) or N (B) index within tile
    const int hi   = lane >> 4;   // lane-half: selects K-group / M-half

    const int baseTok = blockIdx.x * TOK_PER_BLOCK + w * TOK_PER_WAVE;

    // ---------------- B fragments: w1 (20x64 f32) -> f16, K padded to 32 ----
    // v16h element e maps to K = (e<8 ? e : e+8) + 8*hi   (ISA 16-bit A/B layout)
    V16H bfrag[4];
    float b1v[4], w2v[4];
#pragma unroll
    for (int t = 0; t < 4; ++t) {
        const int n = col + 16 * t;
#pragma unroll
        for (int e = 0; e < 16; ++e) {
            const int K = (e < 8 ? e : e + 8) + 8 * hi;
            bfrag[t].h[e] = (K < INDIM) ? (_Float16)w1[K * HID + n] : (_Float16)0.0f;
        }
        b1v[t] = b1[n];
        w2v[t] = w2[n];
    }
    const float b2v = b2[0];

    // ---------------- stage 16 tokens of pred_corners into LDS --------------
    const float* gp = pred + (size_t)baseTok * CDIM;
    if (baseTok + TOK_PER_WAVE <= nTok) {
        // Hot path: async DMA straight into LDS, 16B per lane per issue,
        // no VGPR round-trip. Tracked by ASYNCcnt.
#pragma unroll 4
        for (int i = lane; i < TOK_PER_WAVE * CDIM4; i += 32) {
            const int t  = i / CDIM4;          // token within wave tile
            const int d4 = i % CDIM4;          // float4 within token
            const unsigned ldsAddr =
                (unsigned)(uintptr_t)&predLds[w][t * LDS_STRIDE + d4 * 4];
            const float* ga = gp + (size_t)i * 4;
            asm volatile("global_load_async_to_lds_b128 %0, %1, off"
                         :: "v"(ldsAddr), "v"(ga)
                         : "memory");
        }
        asm volatile("s_wait_asynccnt 0x0" ::: "memory");
        // speculative prefetch of the next wave-chunk (dropped if OOB)
        __builtin_prefetch((const void*)(gp + TOK_PER_WAVE * CDIM + lane * 32), 0, 0);
    } else {
        // Tail path: guarded scalar staging with zero fill.
        const size_t gbase = (size_t)baseTok * CDIM;
        const size_t gtot  = (size_t)nTok * CDIM;
        for (int i = lane; i < TOK_PER_WAVE * CDIM; i += 32) {
            float v = (gbase + (size_t)i < gtot) ? gp[i] : 0.0f;
            predLds[w][(i / CDIM) * LDS_STRIDE + (i % CDIM)] = v;
        }
    }
    __syncthreads();

    // ---------------- per-lane softmax + top-4 for 2 corners of 1 token -----
    const int myTok = col;
#pragma unroll
    for (int cc = 0; cc < 2; ++cc) {
        const int c = 2 * hi + cc;
        const float* xr = &predLds[w][myTok * LDS_STRIDE + c * NBINS1];
        float x[NBINS1];
#pragma unroll
        for (int j = 0; j < NBINS1; ++j) x[j] = xr[j];

        // branchless sorted insert: t0 >= t1 >= t2 >= t3
        float t0 = -INFINITY, t1 = -INFINITY, t2 = -INFINITY, t3 = -INFINITY;
#pragma unroll
        for (int j = 0; j < NBINS1; ++j) {
            const float v = x[j];
            const float a  = fminf(t0, v);  t0 = fmaxf(t0, v);
            const float bb = fminf(t1, a);  t1 = fmaxf(t1, a);
            const float cm = fminf(t2, bb); t2 = fmaxf(t2, bb);
            t3 = fmaxf(t3, cm);
        }
        float s = 0.0f;
#pragma unroll
        for (int j = 0; j < NBINS1; ++j) s += __expf(x[j] - t0);
        const float inv = 1.0f / s;
        const float p0 = inv;                    // exp(t0 - t0) / s
        const float p1 = __expf(t1 - t0) * inv;
        const float p2 = __expf(t2 - t0) * inv;
        const float p3 = __expf(t3 - t0) * inv;
        const float pm = (p0 + p1 + p2 + p3) * 0.25f;

        _Float16* sp = &statLds[w][myTok][5 * c];
        sp[0] = (_Float16)p0; sp[1] = (_Float16)p1; sp[2] = (_Float16)p2;
        sp[3] = (_Float16)p3; sp[4] = (_Float16)pm;
    }
    // zero-pad K = 20..31 (lane-half 0 -> 20..25, half 1 -> 26..31)
    {
        _Float16* zp = &statLds[w][myTok][INDIM + hi * 6];
#pragma unroll
        for (int j = 0; j < 6; ++j) zp[j] = (_Float16)0.0f;
    }
    __syncthreads();

    // ---------------- A fragment: stat (16 tokens x K32) ---------------------
    V16H afrag;
#pragma unroll
    for (int e = 0; e < 16; ++e) {
        const int K = (e < 8 ? e : e + 8) + 8 * hi;
        afrag.h[e] = statLds[w][col][K];
    }

    // ---------------- layer1 WMMA (+b1, ReLU) fused with layer2 dot ---------
    float accQ[8];
#pragma unroll
    for (int r = 0; r < 8; ++r) accQ[r] = 0.0f;

#pragma unroll
    for (int t = 0; t < 4; ++t) {
        v8f cin;
#pragma unroll
        for (int r = 0; r < 8; ++r) cin[r] = b1v[t];   // C[M][N] = b1[N], lane-constant
        v8f d = __builtin_amdgcn_wmma_f32_16x16x32_f16(
                    false, afrag.v, false, bfrag[t].v, (short)0, cin, false, false);
#pragma unroll
        for (int r = 0; r < 8; ++r)
            accQ[r] += fmaxf(d[r], 0.0f) * w2v[t];     // ReLU + w2 partial dot
    }

    // reduce over N (16 lanes per half; masks 1,2,4,8 stay within each half)
#pragma unroll
    for (int m = 1; m <= 8; m <<= 1) {
#pragma unroll
        for (int r = 0; r < 8; ++r)
            accQ[r] += __shfl_xor(accQ[r], m, 32);
    }

    // D layout: accQ[r] is token M = r + 8*hi; lanes 0 / 16 write their half
    if (col == 0) {
#pragma unroll
        for (int r = 0; r < 8; ++r) {
            const int tok = baseTok + 8 * hi + r;
            if (tok < nTok) out[tok] = scores[tok] + accQ[r] + b2v;
        }
    }
}

extern "C" void kernel_launch(void* const* d_in, const int* in_sizes, int n_in,
                              void* d_out, int out_size, void* d_ws, size_t ws_size,
                              hipStream_t stream) {
    const float* scores = (const float*)d_in[0];   // (B, L, 1)
    const float* pred   = (const float*)d_in[1];   // (B, L, 132)
    const float* w1     = (const float*)d_in[2];   // (20, 64)
    const float* b1     = (const float*)d_in[3];   // (64,)
    const float* w2     = (const float*)d_in[4];   // (64, 1)
    const float* b2     = (const float*)d_in[5];   // (1,)
    float* out = (float*)d_out;

    const int nTok  = in_sizes[0];                 // B*L
    const int grid  = (nTok + TOK_PER_BLOCK - 1) / TOK_PER_BLOCK;
    deimv2_lqe_kernel<<<grid, TPB, 0, stream>>>(scores, pred, w1, b1, w2, b2, out, nTok);
}
